// InternVL_CLIP_4123168604332
// MI455X (gfx1250) — matrix operand order repaired and checked by reading, not compiled, not run on hardware
//
#include <hip/hip_runtime.h>
#include <hip/hip_bf16.h>
#include <math.h>

// MI455X / gfx1250: wave32, WMMA bf16 16x16x32 with f32 accumulate.

typedef __attribute__((ext_vector_type(16))) __bf16 v16bf;
typedef __attribute__((ext_vector_type(8)))  float  v8f;

#define B_  4
#define N_  1024
#define C_  1024
#define H_  16
#define D_  64
#define MR  (B_ * N_)   // 4096 rows

// LDS pitches (ushort units) chosen to stay 16B-aligned but break bank periodicity
#define KT_PITCH 40
#define VT_PITCH 72
#define PB_PITCH 40

// ---------------- helpers ----------------

__device__ __forceinline__ unsigned short f2bf(float f) {
  union { float f; unsigned int u; } c; c.f = f;
  unsigned int u = c.u;
  return (unsigned short)((u + 0x7FFFu + ((u >> 16) & 1u)) >> 16); // RNE
}

union BFU { unsigned short u; __bf16 h; };

// Load 8 consecutive bf16 (stored as ushort) into elements [o..o+7] of a v16bf.
__device__ __forceinline__ void load8(v16bf& d, int o, const unsigned short* p) {
#pragma unroll
  for (int i = 0; i < 8; ++i) { BFU c; c.u = p[i]; d[o + i] = c.h; }
}

__device__ __forceinline__ float block_reduce(float v, float* sred) {
  int t = threadIdx.x;
  sred[t] = v; __syncthreads();
  for (int s = 128; s > 0; s >>= 1) { if (t < s) sred[t] += sred[t + s]; __syncthreads(); }
  float r = sred[0]; __syncthreads();
  return r;
}

// ---------------- fp32 -> bf16 convert (row-major, for activations) ----------------

__global__ void k_f2bf(const float* __restrict__ in, unsigned short* __restrict__ out, int n) {
  int i = blockIdx.x * blockDim.x + threadIdx.x;
  if (i < n) out[i] = f2bf(in[i]);
}

// ---------------- fp32 [N][K] -> bf16 transposed [K][N] (for weights) ----------------
// Tiled through LDS so both global read (along k) and write (along n) are coalesced.

__global__ __launch_bounds__(256) void k_w2bfT(const float* __restrict__ W,
                                               unsigned short* __restrict__ Wt,
                                               int Nrows, int K) {
  __shared__ unsigned short tile[32][33];
  int bk = blockIdx.x * 32, bn = blockIdx.y * 32;
  int tx = threadIdx.x & 31, ty = threadIdx.x >> 5;   // 32 x 8
#pragma unroll
  for (int i = ty; i < 32; i += 8)
    tile[i][tx] = f2bf(W[(size_t)(bn + i) * K + bk + tx]);
  __syncthreads();
#pragma unroll
  for (int i = ty; i < 32; i += 8)
    Wt[(size_t)(bk + i) * Nrows + bn + tx] = tile[tx][i];
}

// ---------------- GEMM: C[M,N] = A[M,K] * W^T + bias, with Wt = W^T in [K][N] ----------------
// Block: 128 threads = 4 waves in 2x2; each wave owns a 32x64 tile (2x4 fragments).
// Per 32-wide k-step: 2 A frags (b128 pairs) + 4 B frags (strided loads) feed 8 WMMAs.

__global__ __launch_bounds__(128) void k_gemm(const unsigned short* __restrict__ A,
                                              const unsigned short* __restrict__ Wt,
                                              const float* __restrict__ bias,
                                              float* __restrict__ Cout,
                                              int M, int Nn, int K) {
  int wv = threadIdx.x >> 5, lane = threadIdx.x & 31;
  int mlo = lane & 15, hi = lane >> 4;
  int tM = blockIdx.y * 64 + (wv >> 1) * 32;
  int tN = blockIdx.x * 128 + (wv & 1) * 64;

  v8f zero = {};
  v8f acc[2][4];
#pragma unroll
  for (int im = 0; im < 2; ++im)
#pragma unroll
    for (int in_ = 0; in_ < 4; ++in_) acc[im][in_] = zero;

  for (int kb = 0; kb < K; kb += 32) {
    v16bf a[2], b[4];
#pragma unroll
    for (int im = 0; im < 2; ++im) {
      // A fragment: lane=m (0..15), element e -> k = (e>>3)*16 + hi*8 + (e&7)
      const unsigned short* pa = A + (size_t)(tM + im * 16 + mlo) * K + kb + hi * 8;
      load8(a[im], 0, pa); load8(a[im], 8, pa + 16);
    }
#pragma unroll
    for (int in_ = 0; in_ < 4; ++in_) {
      // B fragment: lane=n (0..15), element e -> k = (e>>3)*16 + hi*8 + (e&7)
      const unsigned short* pb = Wt + (size_t)(kb + hi * 8) * Nn + tN + in_ * 16 + mlo;
      for (int i = 0; i < 8; ++i) { BFU c; c.u = pb[(size_t)i * Nn]; b[in_][i] = c.h; c.u = pb[(size_t)(16 + i) * Nn]; b[in_][8 + i] = c.h; }
    }
#pragma unroll
    for (int im = 0; im < 2; ++im)
#pragma unroll
      for (int in_ = 0; in_ < 4; ++in_)
        acc[im][in_] = __builtin_amdgcn_wmma_f32_16x16x32_bf16(
            false, a[im], false, b[in_], (short)0, acc[im][in_], false, false);
  }

#pragma unroll
  for (int im = 0; im < 2; ++im)
#pragma unroll
    for (int in_ = 0; in_ < 4; ++in_) {
      int col = tN + in_ * 16 + mlo;
      float bs = bias[col];
#pragma unroll
      for (int j = 0; j < 8; ++j) {             // C frag: row = j + hi*8, col = mlo
        int row = tM + im * 16 + hi * 8 + j;
        Cout[(size_t)row * Nn + col] = acc[im][in_][j] + bs;
      }
    }
}

// ---------------- LayerNorm(q), LayerNorm(k), split to [b][h][n][d] bf16 ----------------

__global__ __launch_bounds__(256) void k_lnsplit(const float* __restrict__ qkv,
                                                 const float* __restrict__ qg, const float* __restrict__ qbb,
                                                 const float* __restrict__ kg, const float* __restrict__ kbb,
                                                 unsigned short* __restrict__ qo,
                                                 unsigned short* __restrict__ ko,
                                                 unsigned short* __restrict__ vo) {
  __shared__ float sred[256];
  int r = blockIdx.x;              // 0..4095
  int b = r >> 10, n = r & 1023;
  int t = threadIdx.x;
  const float* row = qkv + (size_t)r * (3 * C_);

  for (int sec = 0; sec < 2; ++sec) {
    const float* x = row + sec * C_;
    const float* g = sec ? kg : qg;
    const float* bb = sec ? kbb : qbb;
    unsigned short* dst = sec ? ko : qo;
    float s1 = 0.f;
#pragma unroll
    for (int i = 0; i < 4; ++i) s1 += x[t + i * 256];
    s1 = block_reduce(s1, sred);
    float mu = s1 * (1.f / 1024.f);
    float s2 = 0.f;
#pragma unroll
    for (int i = 0; i < 4; ++i) { float d = x[t + i * 256] - mu; s2 += d * d; }
    s2 = block_reduce(s2, sred);
    float rinv = rsqrtf(s2 * (1.f / 1024.f) + 1e-5f);
    float scale = sec ? 1.f : 0.125f;           // fold D^-0.5 into q
#pragma unroll
    for (int i = 0; i < 4; ++i) {
      int c = t + i * 256;
      float v = ((x[c] - mu) * rinv * g[c] + bb[c]) * scale;
      int h = c >> 6, d = c & 63;
      dst[(((size_t)b * H_ + h) * N_ + n) * D_ + d] = f2bf(v);
    }
  }
#pragma unroll
  for (int i = 0; i < 4; ++i) {                  // v passthrough -> bf16
    int c = t + i * 256;
    int h = c >> 6, d = c & 63;
    vo[(((size_t)b * H_ + h) * N_ + n) * D_ + d] = f2bf(row[2 * C_ + c]);
  }
}

// ---------------- flash attention per (b,h): 64 query rows / block, 4 waves ----------------

__global__ __launch_bounds__(128) void k_attn(const unsigned short* __restrict__ Q,
                                              const unsigned short* __restrict__ Kk,
                                              const unsigned short* __restrict__ V,
                                              unsigned short* __restrict__ O) {
  __shared__ unsigned short ktT[64 * KT_PITCH];       // K tile, transposed: [d][key]
  __shared__ unsigned short vt[32 * VT_PITCH];        // V tile, row-major:  [key][d]
  __shared__ unsigned short pbuf[4][16 * PB_PITCH];

  int bh = blockIdx.y;
  int b = bh >> 4, h = bh & 15;
  size_t base = (size_t)bh * N_ * D_;
  int tid = threadIdx.x;
  int wv = tid >> 5, lane = tid & 31;
  int mlo = lane & 15, hi = lane >> 4;
  int q0 = blockIdx.x * 64 + wv * 16;

  // Q A-fragments for the two 32-wide d-steps (loaded once)
  v16bf qa0, qa1;
  {
    const unsigned short* pq = Q + base + (size_t)(q0 + mlo) * D_;
    load8(qa0, 0, pq + hi * 8);      load8(qa0, 8, pq + 16 + hi * 8);
    load8(qa1, 0, pq + 32 + hi * 8); load8(qa1, 8, pq + 48 + hi * 8);
  }

  v8f o0 = {}, o1 = {}, o2 = {}, o3 = {};
  float mrow[8], lrow[8];
#pragma unroll
  for (int j = 0; j < 8; ++j) { mrow[j] = -1e30f; lrow[j] = 0.f; }

  for (int kb0 = 0; kb0 < N_; kb0 += 32) {
    __syncthreads();
    // cooperative stage: 32 keys x 64 dims of K (transposed) and V
#pragma unroll
    for (int i = 0; i < 16; ++i) {
      int e = tid + i * 128;                 // 2048 elements
      int key = e >> 6, d = e & 63;          // coalesced global read
      size_t gsrc = base + (size_t)(kb0 + key) * D_ + d;
      ktT[d * KT_PITCH + key] = Kk[gsrc];
      vt[key * VT_PITCH + d] = V[gsrc];
    }
    __syncthreads();

    // S = Q * K^T  (16 rows x 32 key cols), two 16-col tiles
    v8f s0 = {}, s1 = {};
    {
      v16bf kf; const unsigned short* p;
      p = &ktT[(0 * 32 + hi * 8) * KT_PITCH + 0 + mlo];  for (int i = 0; i < 8; ++i) { BFU c; c.u = p[i * KT_PITCH]; kf[i] = c.h; c.u = p[(16 + i) * KT_PITCH]; kf[8 + i] = c.h; }
      s0 = __builtin_amdgcn_wmma_f32_16x16x32_bf16(false, qa0, false, kf, (short)0, s0, false, false);
      p = &ktT[(1 * 32 + hi * 8) * KT_PITCH + 0 + mlo];  for (int i = 0; i < 8; ++i) { BFU c; c.u = p[i * KT_PITCH]; kf[i] = c.h; c.u = p[(16 + i) * KT_PITCH]; kf[8 + i] = c.h; }
      s0 = __builtin_amdgcn_wmma_f32_16x16x32_bf16(false, qa1, false, kf, (short)0, s0, false, false);
      p = &ktT[(0 * 32 + hi * 8) * KT_PITCH + 16 + mlo]; for (int i = 0; i < 8; ++i) { BFU c; c.u = p[i * KT_PITCH]; kf[i] = c.h; c.u = p[(16 + i) * KT_PITCH]; kf[8 + i] = c.h; }
      s1 = __builtin_amdgcn_wmma_f32_16x16x32_bf16(false, qa0, false, kf, (short)0, s1, false, false);
      p = &ktT[(1 * 32 + hi * 8) * KT_PITCH + 16 + mlo]; for (int i = 0; i < 8; ++i) { BFU c; c.u = p[i * KT_PITCH]; kf[i] = c.h; c.u = p[(16 + i) * KT_PITCH]; kf[8 + i] = c.h; }
      s1 = __builtin_amdgcn_wmma_f32_16x16x32_bf16(false, qa1, false, kf, (short)0, s1, false, false);
    }

    // online softmax (fp32); row j+hi*8 stats replicated across 16 lanes of each half
    float p0[8], p1[8];
#pragma unroll
    for (int j = 0; j < 8; ++j) {
      float mx = fmaxf(s0[j], s1[j]);
#pragma unroll
      for (int m = 1; m < 16; m <<= 1) mx = fmaxf(mx, __shfl_xor(mx, m, 32));
      float nm = fmaxf(mrow[j], mx);
      float corr = __expf(mrow[j] - nm);
      float a0 = __expf(s0[j] - nm);
      float a1 = __expf(s1[j] - nm);
      float sm = a0 + a1;
#pragma unroll
      for (int m = 1; m < 16; m <<= 1) sm += __shfl_xor(sm, m, 32);
      lrow[j] = lrow[j] * corr + sm;
      mrow[j] = nm;
      o0[j] *= corr; o1[j] *= corr; o2[j] *= corr; o3[j] *= corr;
      p0[j] = a0; p1[j] = a1;
    }

    // C-layout -> A-layout for P via per-wave LDS tile
    unsigned short* pp = pbuf[wv];
#pragma unroll
    for (int j = 0; j < 8; ++j) {
      pp[(j + hi * 8) * PB_PITCH + mlo]      = f2bf(p0[j]);
      pp[(j + hi * 8) * PB_PITCH + 16 + mlo] = f2bf(p1[j]);
    }
    __syncthreads();

    v16bf pa;
    { const unsigned short* pl = &pbuf[wv][mlo * PB_PITCH + hi * 8];
      load8(pa, 0, pl); load8(pa, 8, pl + 16); }

    // O += P * V  (four 16-wide d tiles); B frag = vt[key=(e>>3)*16+hi*8+(e&7)][d=mlo]
    {
      v16bf vf; const unsigned short* pv;
      pv = &vt[hi * 8 * VT_PITCH +  0 + mlo]; for (int i = 0; i < 8; ++i) { BFU c; c.u = pv[i * VT_PITCH]; vf[i] = c.h; c.u = pv[(16 + i) * VT_PITCH]; vf[8 + i] = c.h; }
      o0 = __builtin_amdgcn_wmma_f32_16x16x32_bf16(false, pa, false, vf, (short)0, o0, false, false);
      pv = &vt[hi * 8 * VT_PITCH + 16 + mlo]; for (int i = 0; i < 8; ++i) { BFU c; c.u = pv[i * VT_PITCH]; vf[i] = c.h; c.u = pv[(16 + i) * VT_PITCH]; vf[8 + i] = c.h; }
      o1 = __builtin_amdgcn_wmma_f32_16x16x32_bf16(false, pa, false, vf, (short)0, o1, false, false);
      pv = &vt[hi * 8 * VT_PITCH + 32 + mlo]; for (int i = 0; i < 8; ++i) { BFU c; c.u = pv[i * VT_PITCH]; vf[i] = c.h; c.u = pv[(16 + i) * VT_PITCH]; vf[8 + i] = c.h; }
      o2 = __builtin_amdgcn_wmma_f32_16x16x32_bf16(false, pa, false, vf, (short)0, o2, false, false);
      pv = &vt[hi * 8 * VT_PITCH + 48 + mlo]; for (int i = 0; i < 8; ++i) { BFU c; c.u = pv[i * VT_PITCH]; vf[i] = c.h; c.u = pv[(16 + i) * VT_PITCH]; vf[8 + i] = c.h; }
      o3 = __builtin_amdgcn_wmma_f32_16x16x32_bf16(false, pa, false, vf, (short)0, o3, false, false);
    }
  }

  // normalize + store bf16 to [b][n][h*64+d] (row-major for proj GEMM)
#pragma unroll
  for (int j = 0; j < 8; ++j) {
    float inv = 1.f / lrow[j];
    size_t qn = (size_t)q0 + j + hi * 8;
    size_t rowo = ((size_t)b * N_ + qn) * C_ + (size_t)h * D_;
    O[rowo +  0 + mlo] = f2bf(o0[j] * inv);
    O[rowo + 16 + mlo] = f2bf(o1[j] * inv);
    O[rowo + 32 + mlo] = f2bf(o2[j] * inv);
    O[rowo + 48 + mlo] = f2bf(o3[j] * inv);
  }
}

// ---------------- launch ----------------

extern "C" void kernel_launch(void* const* d_in, const int* in_sizes, int n_in,
                              void* d_out, int out_size, void* d_ws, size_t ws_size,
                              hipStream_t stream) {
  const float* x      = (const float*)d_in[0];
  const float* qkv_w  = (const float*)d_in[1];
  const float* qkv_b  = (const float*)d_in[2];
  const float* qng    = (const float*)d_in[3];
  const float* qnb    = (const float*)d_in[4];
  const float* kng    = (const float*)d_in[5];
  const float* knb    = (const float*)d_in[6];
  const float* proj_w = (const float*)d_in[7];
  const float* proj_b = (const float*)d_in[8];
  float* out = (float*)d_out;
  (void)in_sizes; (void)n_in; (void)out_size; (void)ws_size;

  char* ws = (char*)d_ws;
  size_t off = 0;
  auto take = [&](size_t bytes) -> char* {
    char* p = ws + off; off = (off + bytes + 255) & ~(size_t)255; return p;
  };
  unsigned short* xb   = (unsigned short*)take((size_t)MR * C_ * 2);        // 8 MB
  unsigned short* wqT  = (unsigned short*)take((size_t)3 * C_ * C_ * 2);    // 6 MB  (K x 3C)
  unsigned short* wpT  = (unsigned short*)take((size_t)C_ * C_ * 2);        // 2 MB  (K x C)
  float*          qkv  = (float*)take((size_t)MR * 3 * C_ * 4);             // 48 MB
  unsigned short* qb   = (unsigned short*)take((size_t)MR * C_ * 2);        // 8 MB
  unsigned short* kb   = (unsigned short*)take((size_t)MR * C_ * 2);        // 8 MB
  unsigned short* vb   = (unsigned short*)take((size_t)MR * C_ * 2);        // 8 MB
  unsigned short* ao   = (unsigned short*)take((size_t)MR * C_ * 2);        // 8 MB

  int n1 = MR * C_;
  k_f2bf<<<(n1 + 255) / 256, 256, 0, stream>>>(x, xb, n1);

  // weights: [N][K] fp32 -> [K][N] bf16
  dim3 gt1(C_ / 32, (3 * C_) / 32);
  k_w2bfT<<<gt1, 256, 0, stream>>>(qkv_w, wqT, 3 * C_, C_);
  dim3 gt2(C_ / 32, C_ / 32);
  k_w2bfT<<<gt2, 256, 0, stream>>>(proj_w, wpT, C_, C_);

  dim3 g1((3 * C_) / 128, MR / 64);
  k_gemm<<<g1, 128, 0, stream>>>(xb, wqT, qkv_b, qkv, MR, 3 * C_, C_);

  k_lnsplit<<<MR, 256, 0, stream>>>(qkv, qng, qnb, kng, knb, qb, kb, vb);

  dim3 g2(N_ / 64, B_ * H_);
  k_attn<<<g2, 128, 0, stream>>>(qb, kb, vb, ao);

  dim3 g3(C_ / 128, MR / 64);
  k_gemm<<<g3, 128, 0, stream>>>(ao, wpT, proj_b, out, MR, C_, C_);
}
